// PoseInitializer_41446434406628
// MI455X (gfx1250) — matrix operand order repaired and checked by reading, not compiled, not run on hardware
//
#include <hip/hip_runtime.h>

// ---------------- types ----------------
typedef __bf16 bf16;
typedef __attribute__((ext_vector_type(16))) __bf16 v16bf;
typedef __attribute__((ext_vector_type(8)))  __bf16 v8bf;
typedef __attribute__((ext_vector_type(8)))  float  v8f;

// ---------------- problem constants ----------------
constexpr int kB  = 4;
constexpr int kN  = 4096;
constexpr int kT  = 8192;
constexpr int kNT = kB * kN;        // 16384 nodes
constexpr int kE  = kB * kT * 6;    // 196608 edges
constexpr int kBT = kB * kT;        // 32768 triangles

__device__ __forceinline__ float siluf(float x) { return x / (1.0f + __expf(-x)); }
__device__ __forceinline__ float sigmf(float x) { return 1.0f / (1.0f + __expf(-x)); }

// monotone float<->uint map for atomic segment-max
__device__ __forceinline__ unsigned fmapu(float f) {
    unsigned u = __float_as_uint(f);
    return (u & 0x80000000u) ? ~u : (u | 0x80000000u);
}
__device__ __forceinline__ float funmap(unsigned u) {
    return __uint_as_float((u & 0x80000000u) ? (u & 0x7fffffffu) : ~u);
}

// ---------------- 3x3 polar decomposition (Newton) : nearest rotation ----------------
__device__ void polar3(const float* Kin, float* R) {
    float X[9];
    float s = 1e-20f;
    for (int i = 0; i < 9; ++i) { X[i] = Kin[i]; s = fmaxf(s, fabsf(Kin[i])); }
    float inv = 1.0f / s;
    for (int i = 0; i < 9; ++i) X[i] *= inv;
    for (int it = 0; it < 12; ++it) {
        float c0 = X[4]*X[8] - X[5]*X[7];
        float c1 = X[5]*X[6] - X[3]*X[8];
        float c2 = X[3]*X[7] - X[4]*X[6];
        float c3 = X[2]*X[7] - X[1]*X[8];
        float c4 = X[0]*X[8] - X[2]*X[6];
        float c5 = X[1]*X[6] - X[0]*X[7];
        float c6 = X[1]*X[5] - X[2]*X[4];
        float c7 = X[2]*X[3] - X[0]*X[5];
        float c8 = X[0]*X[4] - X[1]*X[3];
        float det = X[0]*c0 + X[1]*c1 + X[2]*c2;
        float id  = 1.0f / (det + copysignf(1e-20f, det));
        X[0] = 0.5f*(X[0] + c0*id); X[1] = 0.5f*(X[1] + c1*id); X[2] = 0.5f*(X[2] + c2*id);
        X[3] = 0.5f*(X[3] + c3*id); X[4] = 0.5f*(X[4] + c4*id); X[5] = 0.5f*(X[5] + c5*id);
        X[6] = 0.5f*(X[6] + c6*id); X[7] = 0.5f*(X[7] + c7*id); X[8] = 0.5f*(X[8] + c8*id);
    }
    for (int i = 0; i < 9; ++i) R[i] = X[i];
}

// ============================================================================
// WMMA GEMM with LDS-staged weights:
//   out[M,N] = A[M,K](bf16,row-major) @ WT[N,K](bf16)^T + bias[N]
// block = 256 threads = 8 waves; block owns a 64-wide N panel x 8 M-tiles.
// Weight panel is staged into LDS in K-chunks of 256 with
// global_load_async_to_lds_b128 (ASYNCcnt), then each wave streams A from
// global and pulls B fragments from LDS (ds_load_b128) into
// v_wmma_f32_16x16x32_bf16, K stepped by 32.
// ============================================================================
constexpr int kKC = 256;   // K chunk staged in LDS (64 cols x 256 k x bf16 = 32 KB)

__global__ __launch_bounds__(256)
void wmma_gemm_lds_kernel(const bf16* __restrict__ A, const bf16* __restrict__ WT,
                          const float* __restrict__ bias,
                          float* __restrict__ outF, bf16* __restrict__ outB,
                          int M, int N, int K, int act)
{
    __shared__ bf16 smem[64 * kKC];

    const int tn   = blockIdx.x;              // N/64 panel
    const int wave = threadIdx.x >> 5;        // 0..7
    const int lane = threadIdx.x & 31;
    const int hf   = lane >> 4;               // K-half selector
    const int l16  = lane & 15;
    const int tm   = blockIdx.y * 8 + wave;   // 16-row M tile

    const unsigned lds_base = (unsigned)(uintptr_t)(void*)smem;  // LDS offset in [31:0]

    v8f acc[4] = {};
    const int row = (tm << 4) + l16;
    const bf16* __restrict__ arow = A + (size_t)row * K;

    for (int k0 = 0; k0 < K; k0 += kKC) {
        const int kchunk = (K - k0 < kKC) ? (K - k0) : kKC;
        // ---- cooperative async stage of weight panel chunk: 64 rows x kchunk
        const int upr = kchunk >> 3;                 // 16-byte units per row
        for (int u = threadIdx.x; u < upr * 64; u += 256) {
            const int r  = u / upr;
            const int kk = (u - r * upr) << 3;
            const bf16* src = WT + (size_t)(tn * 64 + r) * K + k0 + kk;
            const unsigned dst = lds_base + (unsigned)(r * kKC + kk) * 2u;
            asm volatile("global_load_async_to_lds_b128 %0, %1, off"
                         :: "v"(dst), "v"(src) : "memory");
        }
        asm volatile("s_wait_asynccnt 0x0" ::: "memory");
        __syncthreads();

        // ---- compute on staged chunk
        for (int kk = 0; kk < kchunk; kk += 32) {
            __builtin_prefetch(arow + k0 + kk + 128, 0, 0);   // global_prefetch_b8
            // A fragment (global): VGPR v<4 hold K = hf*8 + 0..7 ; v>=4 add +16
            v8bf a0 = *(const v8bf*)(arow + k0 + kk + hf * 8);
            v8bf a1 = *(const v8bf*)(arow + k0 + kk + 16 + hf * 8);
            v16bf a = __builtin_shufflevector(a0, a1, 0,1,2,3,4,5,6,7,8,9,10,11,12,13,14,15);
#pragma unroll
            for (int j = 0; j < 4; ++j) {
                // B fragment from LDS: column = j*16 + l16, K = hf*8 + 0..7 ; then +16
                const bf16* wp = &smem[(j * 16 + l16) * kKC + kk + hf * 8];
                v8bf b0 = *(const v8bf*)(wp);
                v8bf b1 = *(const v8bf*)(wp + 16);
                v16bf bm = __builtin_shufflevector(b0, b1, 0,1,2,3,4,5,6,7,8,9,10,11,12,13,14,15);
                acc[j] = __builtin_amdgcn_wmma_f32_16x16x32_bf16(
                    false, a, false, bm, (short)0, acc[j], false, false);
            }
        }
        __syncthreads();
    }

#pragma unroll
    for (int j = 0; j < 4; ++j) {
#pragma unroll
        for (int v = 0; v < 8; ++v) {
            const int m = (tm << 4) + v + (hf << 3);
            const int n = (tn << 6) + (j << 4) + l16;
            float val = acc[j][v] + bias[n];
            if (act == 1) val = siluf(val);
            if (outF) outF[(size_t)m * N + n] = val;
            if (outB) outB[(size_t)m * N + n] = (bf16)val;
        }
    }
}

// ============================================================================
// small helper kernels
// ============================================================================
__global__ void init_kernel(float* gatout, float* den, unsigned* segmax,
                            float* confacc, float* accum)
{
    const long i = (long)blockIdx.x * blockDim.x + threadIdx.x;
    if (i < (long)kNT * 256) gatout[i] = 0.0f;
    if (i < kNT * 4) { den[i] = 0.0f; segmax[i] = fmapu(-3.0e38f); }
    if (i < kNT) confacc[i] = 0.0f;
    if (i < 64)  accum[i] = 0.0f;
}

// transpose+convert weight W[K,N] (f32) -> WT[N,K] (bf16)
__global__ void convert_wT_kernel(const float* __restrict__ W, bf16* __restrict__ WT,
                                  int K, int N)
{
    const int i = blockIdx.x * blockDim.x + threadIdx.x;
    if (i >= K * N) return;
    const int n = i / K, k = i % K;
    WT[(size_t)n * K + k] = (bf16)W[(size_t)k * N + n];
}

// descs (f32, NT x 256) -> first 256 cols of xbf (NT x 320)
__global__ void convert_desc_kernel(const float* __restrict__ descs, bf16* __restrict__ xbf)
{
    const int i = blockIdx.x * blockDim.x + threadIdx.x;
    if (i >= kNT * 256) return;
    const int n = i >> 8, c = i & 255;
    xbf[(size_t)n * 320 + c] = (bf16)descs[i];
}

// positional encoding MLP per node; fills xbf cols 256..319 and norm_uv
__global__ void pos_encode_kernel(const float* __restrict__ kpts, const float* __restrict__ p3d,
                                  const float* __restrict__ w1, const float* __restrict__ b1,
                                  const float* __restrict__ w2, const float* __restrict__ b2,
                                  bf16* __restrict__ xbf, float* __restrict__ nuv)
{
    const int n = blockIdx.x * blockDim.x + threadIdx.x;
    if (n >= kNT) return;
    const float u = kpts[n * 2 + 0] * (1.0f / 1216.0f);
    const float v = kpts[n * 2 + 1] * (1.0f / 352.0f);
    const float d = p3d[n * 3 + 2];
    nuv[n * 2 + 0] = u; nuv[n * 2 + 1] = v;
    float h[32];
#pragma unroll
    for (int j = 0; j < 32; ++j)
        h[j] = siluf(u * w1[j] + v * w1[32 + j] + d * w1[64 + j] + b1[j]);
    for (int j = 0; j < 64; ++j) {
        float acc = b2[j];
#pragma unroll
        for (int i = 0; i < 32; ++i) acc += h[i] * w2[i * 64 + j];
        xbf[(size_t)n * 320 + 256 + j] = (bf16)acc;
    }
}

// edges + edge_attr  (writes directly into d_out regions)
__global__ void edge_build_kernel(const int* __restrict__ tri, const float* __restrict__ nuv,
                                  int* __restrict__ oed, float* __restrict__ oea)
{
    const int e = blockIdx.x * blockDim.x + threadIdx.x;
    if (e >= kE) return;
    const int t = e / 6, j = e % 6;
    const int b = t / kT;
    const int jj = (j < 3) ? j : j - 3;
    const int sl = (j < 3) ? jj : (jj + 1) % 3;
    const int dl = (j < 3) ? (jj + 1) % 3 : jj;
    const int s = tri[t * 3 + sl] + b * kN;
    const int d = tri[t * 3 + dl] + b * kN;
    oed[e] = s; oed[kE + e] = d;
    const float dx = nuv[d * 2 + 0] - nuv[s * 2 + 0];
    const float dy = nuv[d * 2 + 1] - nuv[s * 2 + 1];
    const float dist = sqrtf(dx * dx + dy * dy);
    oea[e * 3 + 0] = dx; oea[e * 3 + 1] = dy; oea[e * 3 + 2] = dist;
}

// attention logits + atomic segment max; one thread per (edge, head)
__global__ void logit_kernel(const int* __restrict__ oed, const float* __restrict__ oea,
                             const float* __restrict__ xl, const float* __restrict__ xr,
                             const float* __restrict__ we, const float* __restrict__ att,
                             float* __restrict__ logit, unsigned* __restrict__ segmax)
{
    const int g = blockIdx.x * blockDim.x + threadIdx.x;
    if (g >= kE * 4) return;
    const int e = g >> 2, h = g & 3;
    const int s = oed[e], d = oed[kE + e];
    const float ea0 = oea[e * 3 + 0], ea1 = oea[e * 3 + 1], ea2 = oea[e * 3 + 2];
    const float* pl = xl + (size_t)s * 256 + h * 64;
    const float* pr = xr + (size_t)d * 256 + h * 64;
    const float* w0 = we + h * 64;
    const float* pa = att + h * 64;
    float lg = 0.0f;
#pragma unroll 8
    for (int dd = 0; dd < 64; ++dd) {
        float eev = ea0 * w0[dd] + ea1 * w0[256 + dd] + ea2 * w0[512 + dd];
        float mm = pl[dd] + pr[dd] + eev;
        mm = (mm > 0.0f) ? mm : 0.2f * mm;      // leaky_relu(0.2)
        lg += mm * pa[dd];
    }
    logit[g] = lg;
    atomicMax(segmax + d * 4 + h, fmapu(lg));
}

// exp(logit - max) + atomic segment sum (logit buffer updated in place)
__global__ void expsum_kernel(const int* __restrict__ oed, float* __restrict__ logit,
                              const unsigned* __restrict__ segmax, float* __restrict__ den)
{
    const int g = blockIdx.x * blockDim.x + threadIdx.x;
    if (g >= kE * 4) return;
    const int e = g >> 2, h = g & 3;
    const int d = oed[kE + e];
    const float ex = __expf(logit[g] - funmap(segmax[d * 4 + h]));
    logit[g] = ex;
    atomicAdd(den + d * 4 + h, ex);
}

// alpha-weighted scatter of xl[src] into gatout[dst]; thread per (edge, channel)
__global__ void scatter_kernel(const int* __restrict__ oed, const float* __restrict__ ex,
                               const float* __restrict__ den, const float* __restrict__ xl,
                               float* __restrict__ gatout)
{
    const long g = (long)blockIdx.x * blockDim.x + threadIdx.x;
    if (g >= (long)kE * 256) return;
    const int e = (int)(g >> 8), c = (int)(g & 255), h = c >> 6;
    const int s = oed[e], d = oed[kE + e];
    const float alpha = ex[e * 4 + h] / (den[d * 4 + h] + 1e-16f);
    atomicAdd(gatout + (size_t)d * 256 + c, alpha * xl[(size_t)s * 256 + c]);
}

// +bias, layernorm(256), silu, +identity -> bf16
__global__ __launch_bounds__(256)
void nodepost_kernel(const float* __restrict__ gatout, const float* __restrict__ gbias,
                     const float* __restrict__ lng, const float* __restrict__ lnb,
                     const float* __restrict__ identity, bf16* __restrict__ outpost)
{
    const int n = blockIdx.x, c = threadIdx.x;
    __shared__ float red[256];
    const float v = gatout[(size_t)n * 256 + c] + gbias[c];
    red[c] = v; __syncthreads();
    for (int s = 128; s > 0; s >>= 1) { if (c < s) red[c] += red[c + s]; __syncthreads(); }
    const float mu = red[0] * (1.0f / 256.0f);
    __syncthreads();
    const float dv = v - mu;
    red[c] = dv * dv; __syncthreads();
    for (int s = 128; s > 0; s >>= 1) { if (c < s) red[c] += red[c + s]; __syncthreads(); }
    const float var = red[0] * (1.0f / 256.0f);
    float o = dv * rsqrtf(var + 1e-5f) * lng[c] + lnb[c];
    o = siluf(o) + identity[(size_t)n * 256 + c];
    outpost[(size_t)n * 256 + c] = (bf16)o;
}

// gather node_feat of 3 triangle vertices -> f_tri (bf16, BT x 768)
__global__ void tri_gather_kernel(const int* __restrict__ tri, const float* __restrict__ nf,
                                  bf16* __restrict__ ftri)
{
    const long g = (long)blockIdx.x * blockDim.x + threadIdx.x;
    if (g >= (long)kBT * 768) return;
    const int tid = (int)(g / 768), c = (int)(g % 768);
    const int vv = c >> 8, cc = c & 255;
    const int b = tid / kT;
    const int node = tri[tid * 3 + vv] + b * kN;
    ftri[g] = (bf16)nf[(size_t)node * 256 + cc];
}

// w = sigmoid(feat2 @ wh_w + wh_b)
__global__ void wh_kernel(const float* __restrict__ feat2, const float* __restrict__ whw,
                          const float* __restrict__ whb, float* __restrict__ wtri)
{
    const int t = blockIdx.x * blockDim.x + threadIdx.x;
    if (t >= kBT) return;
    float s = whb[0];
#pragma unroll 8
    for (int i = 0; i < 128; ++i) s += feat2[(size_t)t * 128 + i] * whw[i];
    wtri[t] = sigmf(s);
}

// geometry stage 1: per-triangle K matrix, rotation, xv, weighted sums
__global__ void geo1_kernel(const int* __restrict__ tri, const float* __restrict__ kp1,
                            const float* __restrict__ p3d, const float* __restrict__ intr,
                            const float* __restrict__ wtri,
                            float* __restrict__ Kmat, float* __restrict__ xv,
                            float* __restrict__ accum)
{
    const int id = blockIdx.x * blockDim.x + threadIdx.x;
    if (id >= kBT) return;
    const int b = id / kT;
    const float fx = intr[b * 4 + 0], fy = intr[b * 4 + 1];
    const float cx = intr[b * 4 + 2], cy = intr[b * 4 + 3];
    float Km[9] = {0,0,0,0,0,0,0,0,0};
#pragma unroll
    for (int v = 0; v < 3; ++v) {
        const int node = b * kN + tri[id * 3 + v];
        const float ux = (kp1[node * 2 + 0] - cx) / fx;
        const float uy = (kp1[node * 2 + 1] - cy) / fy;
        const float q[3] = {ux, uy, 1.0f};
        const float* s3 = p3d + (size_t)node * 3;
#pragma unroll
        for (int i = 0; i < 3; ++i)
#pragma unroll
            for (int j = 0; j < 3; ++j) Km[i * 3 + j] += q[i] * s3[j];
    }
    for (int i = 0; i < 9; ++i) Kmat[(size_t)id * 9 + i] = Km[i];
    float R[9];
    polar3(Km, R);
    const float xvv = fx * (R[2] / (R[8] + 1e-8f)) + cx;
    xv[id] = xvv;
    const float w = wtri[id];
    atomicAdd(&accum[b * 16 + 0], w * xvv);
    atomicAdd(&accum[b * 16 + 1], w);
}

// geometry stage 2: gaussian reweighting sums
__global__ void geo2_kernel(const float* __restrict__ xv, const float* __restrict__ wtri,
                            float* __restrict__ accum)
{
    const int id = blockIdx.x * blockDim.x + threadIdx.x;
    if (id >= kBT) return;
    const int b = id / kT;
    const float mu0 = accum[b * 16 + 0] / (accum[b * 16 + 1] + 1e-12f);
    const float xvv = xv[id];
    const float kk = __expf(-(xvv - mu0) * (xvv - mu0) * (1.0f / 8.0f));  // 2*SIGMA^2=8
    const float w = wtri[id];
    atomicAdd(&accum[b * 16 + 2], w * kk * xvv);
    atomicAdd(&accum[b * 16 + 3], w * kk);
}

// geometry stage 3: cw output, Hm accumulation, conf scatter
__global__ void geo3_kernel(const int* __restrict__ tri, const float* __restrict__ xv,
                            const float* __restrict__ wtri, const float* __restrict__ Kmat,
                            float* __restrict__ accum, float* __restrict__ ocw,
                            float* __restrict__ confacc)
{
    const int id = blockIdx.x * blockDim.x + threadIdx.x;
    if (id >= kBT) return;
    const int b = id / kT;
    const float xstar = accum[b * 16 + 2] / (accum[b * 16 + 3] + 1e-12f);
    const float xvv = xv[id];
    const float sj = __expf(-(xvv - xstar) * (xvv - xstar) * (1.0f / 8.0f));
    const float cwv = wtri[id] * sj;
    ocw[id] = cwv;
#pragma unroll
    for (int i = 0; i < 9; ++i)
        atomicAdd(&accum[b * 16 + 4 + i], cwv * Kmat[(size_t)id * 9 + i]);
#pragma unroll
    for (int v = 0; v < 3; ++v)
        atomicAdd(&confacc[b * kN + tri[id * 3 + v]], sj);
}

// geometry stage 4: final R per batch
__global__ void geo4_kernel(const float* __restrict__ accum, float* __restrict__ oR)
{
    const int b = threadIdx.x;
    if (b >= kB) return;
    float Hm[9], R[9];
    for (int i = 0; i < 9; ++i) Hm[i] = accum[b * 16 + 4 + i];
    polar3(Hm, R);
    for (int i = 0; i < 9; ++i) oR[b * 9 + i] = R[i];
}

__global__ void conf_kernel(const float* __restrict__ confacc, float* __restrict__ oconf)
{
    const int n = blockIdx.x * blockDim.x + threadIdx.x;
    if (n < kNT) oconf[n] = tanhf(confacc[n]);
}

// ============================================================================
// host launcher
// ============================================================================
extern "C" void kernel_launch(void* const* d_in, const int* in_sizes, int n_in,
                              void* d_out, int out_size, void* d_ws, size_t ws_size,
                              hipStream_t stream)
{
    // ---- inputs (setup_inputs dict order, params expanded in insertion order)
    const float* descs   = (const float*)d_in[0];
    const float* kpts    = (const float*)d_in[1];
    const float* pts3d   = (const float*)d_in[2];
    const int*   tri     = (const int*)  d_in[3];
    const float* kpts1   = (const float*)d_in[4];
    const float* intr    = (const float*)d_in[5];
    const float* pe_w1   = (const float*)d_in[6];
    const float* pe_b1   = (const float*)d_in[7];
    const float* pe_w2   = (const float*)d_in[8];
    const float* pe_b2   = (const float*)d_in[9];
    const float* res_w   = (const float*)d_in[10];
    const float* res_b   = (const float*)d_in[11];
    const float* gat_wl  = (const float*)d_in[12];
    const float* gat_bl  = (const float*)d_in[13];
    const float* gat_wr  = (const float*)d_in[14];
    const float* gat_br  = (const float*)d_in[15];
    const float* gat_we  = (const float*)d_in[16];
    const float* gat_att = (const float*)d_in[17];
    const float* gat_bias= (const float*)d_in[18];
    const float* ln_g    = (const float*)d_in[19];
    const float* ln_b    = (const float*)d_in[20];
    const float* proj_w  = (const float*)d_in[21];
    const float* proj_b  = (const float*)d_in[22];
    const float* tri_w1  = (const float*)d_in[23];
    const float* tri_b1  = (const float*)d_in[24];
    const float* tri_w2  = (const float*)d_in[25];
    const float* tri_b2  = (const float*)d_in[26];
    const float* wh_w    = (const float*)d_in[27];
    const float* wh_b    = (const float*)d_in[28];

    // ---- outputs: R(36) | cw(32768) | conf(16384) | edges(393216 i32) | edge_attr(589824)
    float* out    = (float*)d_out;
    float* oR     = out;
    float* ocw    = out + 36;
    float* oconf  = out + 36 + 32768;
    int*   oedges = (int*)(out + 49188);
    float* oeattr = out + 49188 + 2 * kE;

    // ---- workspace carve-out
    char* wp = (char*)d_ws;
    auto alloc = [&](size_t bytes) -> char* {
        char* p = wp; wp += (bytes + 255) & ~(size_t)255; return p;
    };
    bf16*  xbf      = (bf16*) alloc((size_t)kNT * 320 * 2);
    float* nuv      = (float*)alloc((size_t)kNT * 2 * 4);
    bf16*  resWT    = (bf16*) alloc((size_t)256 * 320 * 2);
    bf16*  wlT      = (bf16*) alloc((size_t)256 * 320 * 2);
    bf16*  wrT      = (bf16*) alloc((size_t)256 * 320 * 2);
    bf16*  projWT   = (bf16*) alloc((size_t)256 * 256 * 2);
    bf16*  tw1T     = (bf16*) alloc((size_t)128 * 768 * 2);
    bf16*  tw2T     = (bf16*) alloc((size_t)128 * 128 * 2);
    float* identity = (float*)alloc((size_t)kNT * 256 * 4);
    float* xl       = (float*)alloc((size_t)kNT * 256 * 4);
    float* xr       = (float*)alloc((size_t)kNT * 256 * 4);
    float* logit    = (float*)alloc((size_t)kE * 4 * 4);     // becomes exp() in-place
    unsigned* segmx = (unsigned*)alloc((size_t)kNT * 4 * 4);
    float* den      = (float*)alloc((size_t)kNT * 4 * 4);
    float* gatout   = (float*)alloc((size_t)kNT * 256 * 4);
    bf16*  outpost  = (bf16*) alloc((size_t)kNT * 256 * 2);
    float* nodefeat = (float*)alloc((size_t)kNT * 256 * 4);
    bf16*  ftri     = (bf16*) alloc((size_t)kBT * 768 * 2);
    bf16*  feat1    = (bf16*) alloc((size_t)kBT * 128 * 2);
    float* feat2    = (float*)alloc((size_t)kBT * 128 * 4);
    float* wtri     = (float*)alloc((size_t)kBT * 4);
    float* Kmat     = (float*)alloc((size_t)kBT * 9 * 4);
    float* xv       = (float*)alloc((size_t)kBT * 4);
    float* accum    = (float*)alloc((size_t)kB * 16 * 4);
    float* confacc  = (float*)alloc((size_t)kNT * 4);

    auto cdiv = [](long a, long b) { return (int)((a + b - 1) / b); };

    // 0) init accumulators
    init_kernel<<<cdiv((long)kNT * 256, 256), 256, 0, stream>>>(gatout, den, segmx, confacc, accum);

    // 1) build bf16 node features x = [descs | pos_feat]
    convert_desc_kernel<<<cdiv((long)kNT * 256, 256), 256, 0, stream>>>(descs, xbf);
    pos_encode_kernel<<<cdiv(kNT, 128), 128, 0, stream>>>(kpts, pts3d, pe_w1, pe_b1, pe_w2, pe_b2, xbf, nuv);

    // 2) transpose+convert weights to bf16
    convert_wT_kernel<<<cdiv(320 * 256, 256), 256, 0, stream>>>(res_w,  resWT, 320, 256);
    convert_wT_kernel<<<cdiv(320 * 256, 256), 256, 0, stream>>>(gat_wl, wlT,   320, 256);
    convert_wT_kernel<<<cdiv(320 * 256, 256), 256, 0, stream>>>(gat_wr, wrT,   320, 256);
    convert_wT_kernel<<<cdiv(256 * 256, 256), 256, 0, stream>>>(proj_w, projWT, 256, 256);
    convert_wT_kernel<<<cdiv(768 * 128, 256), 256, 0, stream>>>(tri_w1, tw1T,  768, 128);
    convert_wT_kernel<<<cdiv(128 * 128, 256), 256, 0, stream>>>(tri_w2, tw2T,  128, 128);

    // 3) WMMA GEMMs (LDS-staged weights): grid = (N/64, M/128), 256 threads
    auto gemm = [&](const bf16* A, const bf16* WT, const float* bias,
                    float* oF, bf16* oB, int M, int N, int K, int act) {
        wmma_gemm_lds_kernel<<<dim3(N >> 6, M >> 7), 256, 0, stream>>>(
            A, WT, bias, oF, oB, M, N, K, act);
    };
    gemm(xbf, resWT, res_b,  identity, nullptr, kNT, 256, 320, 0);
    gemm(xbf, wlT,   gat_bl, xl,       nullptr, kNT, 256, 320, 0);
    gemm(xbf, wrT,   gat_br, xr,       nullptr, kNT, 256, 320, 0);

    // 4) edges + GAT segment softmax + scatter aggregation
    edge_build_kernel<<<cdiv(kE, 256), 256, 0, stream>>>(tri, nuv, oedges, oeattr);
    logit_kernel<<<cdiv((long)kE * 4, 256), 256, 0, stream>>>(oedges, oeattr, xl, xr, gat_we, gat_att, logit, segmx);
    expsum_kernel<<<cdiv((long)kE * 4, 256), 256, 0, stream>>>(oedges, logit, segmx, den);
    scatter_kernel<<<cdiv((long)kE * 256, 256), 256, 0, stream>>>(oedges, logit, den, xl, gatout);

    // 5) layernorm + silu + residual, then projection GEMM (M=16384, N=256, K=256)
    nodepost_kernel<<<kNT, 256, 0, stream>>>(gatout, gat_bias, ln_g, ln_b, identity, outpost);
    gemm(outpost, projWT, proj_b, nodefeat, nullptr, kNT, 256, 256, 0);

    // 6) triangle MLP (M=32768): 768->128 silu, 128->128 silu, 128->1 sigmoid
    tri_gather_kernel<<<cdiv((long)kBT * 768, 256), 256, 0, stream>>>(tri, nodefeat, ftri);
    gemm(ftri,  tw1T, tri_b1, nullptr, feat1, kBT, 128, 768, 1);
    gemm(feat1, tw2T, tri_b2, feat2, nullptr, kBT, 128, 128, 1);
    wh_kernel<<<cdiv(kBT, 256), 256, 0, stream>>>(feat2, wh_w, wh_b, wtri);

    // 7) per-batch geometry: K matrices, rotations, robust weighting, outputs
    geo1_kernel<<<cdiv(kBT, 128), 128, 0, stream>>>(tri, kpts1, pts3d, intr, wtri, Kmat, xv, accum);
    geo2_kernel<<<cdiv(kBT, 256), 256, 0, stream>>>(xv, wtri, accum);
    geo3_kernel<<<cdiv(kBT, 256), 256, 0, stream>>>(tri, xv, wtri, Kmat, accum, ocw, confacc);
    geo4_kernel<<<1, 32, 0, stream>>>(accum, oR);
    conf_kernel<<<cdiv(kNT, 256), 256, 0, stream>>>(confacc, oconf);

    (void)in_sizes; (void)n_in; (void)out_size; (void)ws_size;
}
